// SparseTF_15504831938689
// MI455X (gfx1250) — matrix operand order repaired and checked by reading, not compiled, not run on hardware
//
#include <hip/hip_runtime.h>

typedef __attribute__((ext_vector_type(16))) _Float16 v16h;
typedef __attribute__((ext_vector_type(8)))  _Float16 v8h;
typedef __attribute__((ext_vector_type(8)))  float    v8f;

#define INPUT_DIM 8192
#define UNITS     4096
#define BATCH     4096
#define NNZ       400000

#define BM 128
#define BN 128
#define BK 64
#define LDSK 80                      // padded LDS row stride in halfs (160 B)
#define LDS_BUF_BYTES (BM * LDSK * 2)  // 20480 B per tile buffer

// ---------------- zero f16 buffer (8 halfs per thread) ----------------
__global__ void zero_f16(uint4* __restrict__ p, int n16) {
  int i = blockIdx.x * blockDim.x + threadIdx.x;
  if (i < n16) {
    uint4 z; z.x = 0u; z.y = 0u; z.z = 0u; z.w = 0u;
    p[i] = z;
  }
}

// ---------------- convert f32 -> f16, 8 elements per thread ----------------
__global__ void f32_to_f16_x8(const float* __restrict__ src, _Float16* __restrict__ dst) {
  long i = (long)(blockIdx.x * blockDim.x + threadIdx.x) * 8;
  float4 a = *(const float4*)(src + i);
  float4 b = *(const float4*)(src + i + 4);
  v8h h = { (_Float16)a.x, (_Float16)a.y, (_Float16)a.z, (_Float16)a.w,
            (_Float16)b.x, (_Float16)b.y, (_Float16)b.z, (_Float16)b.w };
  *(v8h*)(dst + i) = h;
}

// ---------------- scatter-add into transposed f16 kernel Bt[c][r] ----------------
__global__ void scatter_f16(const int* __restrict__ ind, const float* __restrict__ vals,
                            _Float16* __restrict__ Bt, int nnz) {
  int i = blockIdx.x * blockDim.x + threadIdx.x;
  if (i >= nnz) return;
  int r = ind[2 * i];
  int c = ind[2 * i + 1];
  size_t idx = (size_t)c * INPUT_DIM + (size_t)r;   // transposed: Bt[c][r]
  float v = vals[i];
  unsigned* w = (unsigned*)((uintptr_t)(Bt + idx) & ~(uintptr_t)3);
  int sel = (int)(idx & 1);
  unsigned old = *w;
  for (;;) {
    union { unsigned u; _Float16 h[2]; } cur, nxt;
    cur.u = old;
    nxt.u = cur.u;
    nxt.h[sel] = (_Float16)((float)cur.h[sel] + v);
    unsigned prev = atomicCAS(w, old, nxt.u);
    if (prev == old) break;
    old = prev;
  }
}

// ---------------- async global -> LDS, GVS mode ----------------
// saddr = uniform 64-bit base (SGPR pair), vaddr = per-lane u32 byte offset,
// vdst = LDS byte address. Tracked by ASYNCcnt; no data VGPRs.
__device__ __forceinline__ void async_cp16(const void* sbase, unsigned goff, unsigned ldsaddr) {
  asm volatile("global_load_async_to_lds_b128 %0, %1, %2"
               :: "v"(ldsaddr), "v"(goff), "s"(sbase)
               : "memory");
}

// ---------------- GEMM: out = tanh(A @ Bt^T + bias) ----------------
// A  : [BATCH][INPUT_DIM] f16 (K contiguous), Bt : [UNITS][INPUT_DIM] f16 (K contiguous)
// 128x128 block tile, BK=64, 8 waves (4M x 2N), each wave 32x64 via 2x4 WMMA tiles.
// Double-buffered LDS fed by GLOBAL_LOAD_ASYNC_TO_LDS_B128 (8 chunks/thread/tile).
__global__ __launch_bounds__(256) void gemm_bias_tanh(
    const _Float16* __restrict__ A, const _Float16* __restrict__ Bt,
    const float* __restrict__ bias, float* __restrict__ out) {
  __shared__ _Float16 As[2][BM * LDSK];
  __shared__ _Float16 Bs[2][BN * LDSK];

  const int tid  = threadIdx.x;
  const int lane = tid & 31;
  const int wave = tid >> 5;     // 0..7
  const int wm   = wave >> 1;    // 0..3 -> M offset wm*32
  const int wn   = wave & 1;     // 0..1 -> N offset wn*64
  const int lr   = lane & 15;
  const int lh   = lane >> 4;

  const int m0 = blockIdx.y * BM;
  const int n0 = blockIdx.x * BN;

  v8f acc[2][4];
#pragma unroll
  for (int mi = 0; mi < 2; ++mi)
#pragma unroll
    for (int ni = 0; ni < 4; ++ni)
      acc[mi][ni] = (v8f){0.f, 0.f, 0.f, 0.f, 0.f, 0.f, 0.f, 0.f};

  // Copy mapping: tile = 128 rows x 64 halfs = 1024 x 16B chunks; 4 chunks/thread.
  // u32 byte offsets (buffers are 64 MB) + u32 LDS addresses for buffer 0.
  unsigned ga[4], gb[4], sl[4];
#pragma unroll
  for (int i = 0; i < 4; ++i) {
    int chunk = tid + 256 * i;
    int row = chunk >> 3;
    int cc  = chunk & 7;
    ga[i] = (unsigned)(((m0 + row) * INPUT_DIM + cc * 8) * 2);  // bytes
    gb[i] = (unsigned)(((n0 + row) * INPUT_DIM + cc * 8) * 2);  // bytes
    sl[i] = (unsigned)(uintptr_t)&As[0][row * LDSK + cc * 8];   // LDS byte addr (buf 0)
  }
  const unsigned sAB = (unsigned)((uintptr_t)&Bs[0][0] - (uintptr_t)&As[0][0]);

  const int NK = INPUT_DIM / BK;           // 128 k-tiles
  const unsigned KSTEP_B = BK * 2;         // 128 bytes per k-tile

  // Prologue: issue tiles 0 and 1 (8 async ops per thread per tile).
#pragma unroll
  for (int i = 0; i < 4; ++i) {
    async_cp16(A,  ga[i], sl[i]);
    async_cp16(Bt, gb[i], sl[i] + sAB);
  }
#pragma unroll
  for (int i = 0; i < 4; ++i) {
    async_cp16(A,  ga[i] + KSTEP_B, sl[i] + LDS_BUF_BYTES);
    async_cp16(Bt, gb[i] + KSTEP_B, sl[i] + sAB + LDS_BUF_BYTES);
  }

  for (int kt = 0; kt < NK; ++kt) {
    const int cur = kt & 1;
    // Retire this tile's 8 ops; the next tile's 8 may remain in flight.
    if (kt + 1 < NK) asm volatile("s_wait_asynccnt 0x8" ::: "memory");
    else             asm volatile("s_wait_asynccnt 0x0" ::: "memory");
    __syncthreads();   // all waves' DMA data visible

#pragma unroll
    for (int ks = 0; ks < 2; ++ks) {      // two 16x16x32 k-steps per BK=64 tile
      const int kk = ks * 32;
      v16h af[2], bf[4];
      // A frag (16x32 f16): lanes 0-15 hold K{0..7,16..23}; lanes 16-31 K{8..15,24..31}
#pragma unroll
      for (int mi = 0; mi < 2; ++mi) {
        const _Float16* p = &As[cur][(wm * 32 + mi * 16 + lr) * LDSK + kk + lh * 8];
        v8h lo = *(const v8h*)p;
        v8h hi = *(const v8h*)(p + 16);
        af[mi] = __builtin_shufflevector(lo, hi, 0,1,2,3,4,5,6,7,8,9,10,11,12,13,14,15);
      }
      // B frag (32x16 f16): lane holds column n=lr, lanes 0-15 K{0..7,16..23}; lanes 16-31 K{8..15,24..31}
#pragma unroll
      for (int ni = 0; ni < 4; ++ni) {
        const _Float16* p = &Bs[cur][(wn * 64 + ni * 16 + lr) * LDSK + kk + lh * 8];
        v8h lo = *(const v8h*)p;
        v8h hi = *(const v8h*)(p + 16);
        bf[ni] = __builtin_shufflevector(lo, hi, 0,1,2,3,4,5,6,7,8,9,10,11,12,13,14,15);
      }
#pragma unroll
      for (int mi = 0; mi < 2; ++mi)
#pragma unroll
        for (int ni = 0; ni < 4; ++ni)
          acc[mi][ni] = __builtin_amdgcn_wmma_f32_16x16x32_f16(
              false, af[mi], false, bf[ni], (short)0, acc[mi][ni], false, false);
    }

    __syncthreads();   // all waves done reading buf[cur] before DMA overwrites it
    if (kt + 2 < NK) {
      const unsigned kadd = (unsigned)(kt + 2) * KSTEP_B;
      const unsigned sbuf = (unsigned)cur * LDS_BUF_BYTES;
#pragma unroll
      for (int i = 0; i < 4; ++i) {
        async_cp16(A,  ga[i] + kadd, sl[i] + sbuf);
        async_cp16(Bt, gb[i] + kadd, sl[i] + sAB + sbuf);
      }
    }
  }

  // Epilogue: C tile layout -> VGPR j: M = j + 8*lh, N = lr.
#pragma unroll
  for (int ni = 0; ni < 4; ++ni) {
    const int ncol = n0 + wn * 64 + ni * 16 + lr;
    const float bv = bias[ncol];
#pragma unroll
    for (int mi = 0; mi < 2; ++mi) {
      const int mrow = m0 + wm * 32 + mi * 16 + lh * 8;
#pragma unroll
      for (int j = 0; j < 8; ++j) {
        out[(size_t)(mrow + j) * UNITS + ncol] = tanhf(acc[mi][ni][j] + bv);
      }
    }
  }
}

extern "C" void kernel_launch(void* const* d_in, const int* in_sizes, int n_in,
                              void* d_out, int out_size, void* d_ws, size_t ws_size,
                              hipStream_t stream) {
  const float* inputs = (const float*)d_in[0];   // [BATCH, INPUT_DIM] f32
  const int*   ind    = (const int*)d_in[1];     // [NNZ, 2] i32 (row, col)
  const float* kvals  = (const float*)d_in[2];   // [NNZ] f32
  const float* bias   = (const float*)d_in[3];   // [UNITS] f32
  float* out = (float*)d_out;                    // [BATCH, UNITS] f32

  // Workspace: Ah (64 MB) + Bth (64 MB) = 128 MB
  _Float16* Ah  = (_Float16*)d_ws;
  _Float16* Bth = Ah + (size_t)BATCH * INPUT_DIM;

  {
    int n16 = (UNITS * INPUT_DIM) / 8;
    zero_f16<<<n16 / 256, 256, 0, stream>>>((uint4*)Bth, n16);
  }
  {
    long nthreads = ((long)BATCH * INPUT_DIM) / 8;
    f32_to_f16_x8<<<(int)(nthreads / 256), 256, 0, stream>>>(inputs, Ah);
  }
  scatter_f16<<<(NNZ + 255) / 256, 256, 0, stream>>>(ind, kvals, Bth, NNZ);
  {
    dim3 grid(UNITS / BN, BATCH / BM);
    gemm_bias_tanh<<<grid, 256, 0, stream>>>(Ah, Bth, bias, out);
  }
}
